// RBFLayer_38800734552376
// MI455X (gfx1250) — matrix operand order repaired and checked by reading, not compiled, not run on hardware
//
#include <hip/hip_runtime.h>
#include <hip/hip_bf16.h>

// RBF layer: out = exp(-GAMMA * (||x||^2 - 2 x.mu + ||mu||^2))
// BATCH=16384, DIM=512, UNITS=1024, fp32 in/out.
// Split-bf16 WMMA GEMM (hi/lo decomposition, 3x v_wmma_f32_16x16x32_bf16 per
// K=32 chunk) + fused exp epilogue. Tile fills via GLOBAL_LOAD_ASYNC_TO_LDS
// (ASYNCcnt) software pipeline; fp32->bf16 conversion done LDS->LDS.

#define GAMMA 0.5f
#define BATCH 16384
#define DIM   512
#define UNITS 1024

typedef __attribute__((ext_vector_type(4)))  __bf16 bf16x4;
typedef __attribute__((ext_vector_type(8)))  __bf16 bf16x8;
typedef __attribute__((ext_vector_type(16))) __bf16 bf16x16;
typedef __attribute__((ext_vector_type(8)))  float  f32x8;

#define LDS_STRIDE 40   // bf16 tile row stride: 32 data + 8 pad (80B, 16B-aligned)
#define XRAW_STRIDE 36  // raw X fp32 row stride (144B, 16B-aligned)
#define MRAW_STRIDE 132 // raw MU fp32 row stride (528B, 16B-aligned)

#ifndef __has_builtin
#define __has_builtin(x) 0
#endif
#if __has_builtin(__builtin_amdgcn_global_load_async_to_lds_b128)
#define USE_ASYNC 1
#else
#define USE_ASYNC 0
#endif

// The async builtin's params are pointers to 16B int vectors (per hipcc
// diagnostic: "int __vector(4) __device__ *"). Use GNU vector spelling to
// match the builtin's pointee type exactly.
typedef int gv4i __attribute__((vector_size(16)));
typedef __attribute__((address_space(1))) gv4i* as1_v4i_ptr;
typedef __attribute__((address_space(3))) gv4i* as3_v4i_ptr;

__device__ __forceinline__ void async_copy16(const float* g, float* l) {
#if USE_ASYNC
  // Generic global addresses == AS1 addresses; generic LDS addresses carry the
  // LDS byte offset in the low 32 bits (ISA aperture mapping).
  __builtin_amdgcn_global_load_async_to_lds_b128(
      (as1_v4i_ptr)(uintptr_t)g,
      (as3_v4i_ptr)(uint32_t)(uintptr_t)l,
      /*offset=*/0, /*cpol=*/0);
#else
  *(float4*)l = *(const float4*)g;  // sync fallback
#endif
}

__device__ __forceinline__ void wait_async() {
#if USE_ASYNC
#if __has_builtin(__builtin_amdgcn_s_wait_asynccnt)
  __builtin_amdgcn_s_wait_asynccnt(0);
#else
  asm volatile("s_wait_asynccnt 0x0" ::: "memory");
#endif
#endif
}

__device__ __forceinline__ bf16x16 concat8(bf16x8 a, bf16x8 b) {
  return __builtin_shufflevector(a, b, 0,1,2,3,4,5,6,7,8,9,10,11,12,13,14,15);
}

__device__ __forceinline__ f32x8 wmma_bf16(bf16x16 a, bf16x16 b, f32x8 c) {
  // (neg_a, A, neg_b, B, c_mod, C, reuse_a, reuse_b)
  return __builtin_amdgcn_wmma_f32_16x16x32_bf16(false, a, false, b, (short)0, c,
                                                 false, false);
}

// ---------------------------------------------------------------------------
// x_sq[row] = sum_k X[row][k]^2 ; one wave32 per row
// ---------------------------------------------------------------------------
__global__ __launch_bounds__(256) void xsq_kernel(const float* __restrict__ X,
                                                  float* __restrict__ xsq) {
  const int lane = threadIdx.x & 31;
  const int row  = (blockIdx.x * blockDim.x + threadIdx.x) >> 5;
  if (row >= BATCH) return;
  const float* p = X + (size_t)row * DIM;
  float s = 0.0f;
  #pragma unroll
  for (int i = 0; i < DIM / 32; ++i) {
    float v = p[lane + 32 * i];
    s += v * v;
  }
  #pragma unroll
  for (int off = 16; off > 0; off >>= 1) s += __shfl_down(s, off, 32);
  if (lane == 0) xsq[row] = s;
}

// ---------------------------------------------------------------------------
// mu_sq[u] = sum_k MU[k][u]^2
// ---------------------------------------------------------------------------
__global__ __launch_bounds__(256) void musq_kernel(const float* __restrict__ MU,
                                                   float* __restrict__ musq) {
  const int u = blockIdx.x * blockDim.x + threadIdx.x;
  if (u >= UNITS) return;
  float s = 0.0f;
  #pragma unroll 8
  for (int k = 0; k < DIM; ++k) {
    float v = MU[(size_t)k * UNITS + u];
    s += v * v;
  }
  musq[u] = s;
}

// ---------------------------------------------------------------------------
// Main fused kernel: 128x128 output tile per workgroup, 8 waves, each wave a
// 32x64 region (2x4 WMMA 16x16 tiles). Async-pipelined tile fills.
// ---------------------------------------------------------------------------
__global__ __launch_bounds__(256) void rbf_wmma_kernel(
    const float* __restrict__ X, const float* __restrict__ MU,
    const float* __restrict__ xsq, const float* __restrict__ musq,
    float* __restrict__ out) {
  // bf16 compute tiles: X row-major [m][k]; MU transposed [n][k]
  __shared__ __align__(16) __bf16 Xhi[128][LDS_STRIDE];
  __shared__ __align__(16) __bf16 Xlo[128][LDS_STRIDE];
  __shared__ __align__(16) __bf16 Mhi[128][LDS_STRIDE];
  __shared__ __align__(16) __bf16 Mlo[128][LDS_STRIDE];
  // raw fp32 staging tiles (async DMA targets)
  __shared__ __align__(16) float Xraw[128 * XRAW_STRIDE];
  __shared__ __align__(16) float Mraw[32 * MRAW_STRIDE];

  const int tid  = threadIdx.x;
  const int lane = tid & 31;
  const int wave = tid >> 5;
  const int wm   = wave & 3;   // 4 waves along M: 32 rows each
  const int wn   = wave >> 2;  // 2 waves along N: 64 cols each
  const int r    = lane & 15;
  const int h    = lane >> 4;  // lane half selects K-group per ISA layout

  const int n0 = blockIdx.x * 128;
  const int m0 = blockIdx.y * 128;

  // ---- issue async fill of raw tiles for K-chunk kc ----
  auto issue_fill = [&](int kc) {
    #pragma unroll
    for (int i = 0; i < 4; ++i) {
      int idx = tid + 256 * i;          // 1024 x float4 each
      // X tile: 128 rows x 32 floats
      int rr = idx >> 3;
      int c4 = (idx & 7) << 2;
      async_copy16(X + (size_t)(m0 + rr) * DIM + kc + c4,
                   &Xraw[rr * XRAW_STRIDE + c4]);
      // MU tile: 32 K-rows x 128 floats
      int kk = idx >> 5;
      int d4 = (idx & 31) << 2;
      async_copy16(MU + (size_t)(kc + kk) * UNITS + n0 + d4,
                   &Mraw[kk * MRAW_STRIDE + d4]);
    }
  };

  // ---- convert raw fp32 tiles -> bf16 hi/lo compute tiles ----
  auto convert_tiles = [&]() {
    #pragma unroll
    for (int i = 0; i < 4; ++i) {
      int idx = tid + 256 * i;
      // X: row-major, vector stores
      int rr = idx >> 3;
      int c4 = (idx & 7) << 2;
      float4 v = *(const float4*)&Xraw[rr * XRAW_STRIDE + c4];
      const float* vf = reinterpret_cast<const float*>(&v);
      bf16x4 hv, lv;
      #pragma unroll
      for (int j = 0; j < 4; ++j) {
        float  f  = vf[j];
        __bf16 hb = (__bf16)f;
        hv[j] = hb;
        lv[j] = (__bf16)(f - (float)hb);
      }
      *(bf16x4*)&Xhi[rr][c4] = hv;
      *(bf16x4*)&Xlo[rr][c4] = lv;
      // MU: transpose into [n][k], scattered b16 stores
      int kk = idx >> 5;
      int d4 = (idx & 31) << 2;
      float4 w = *(const float4*)&Mraw[kk * MRAW_STRIDE + d4];
      const float* wf = reinterpret_cast<const float*>(&w);
      #pragma unroll
      for (int j = 0; j < 4; ++j) {
        float  f  = wf[j];
        __bf16 hb = (__bf16)f;
        Mhi[d4 + j][kk] = hb;
        Mlo[d4 + j][kk] = (__bf16)(f - (float)hb);
      }
    }
  };

  f32x8 acc[2][4];
  #pragma unroll
  for (int mi = 0; mi < 2; ++mi)
    #pragma unroll
    for (int ni = 0; ni < 4; ++ni)
      #pragma unroll
      for (int j = 0; j < 8; ++j) acc[mi][ni][j] = 0.0f;

  // ---- pipeline prologue ----
  issue_fill(0);
  wait_async();
  __syncthreads();
  convert_tiles();
  __syncthreads();

  // ---- main K loop ----
  for (int kc = 0; kc < DIM; kc += 32) {
    const bool more = (kc + 32) < DIM;
    if (more) issue_fill(kc + 32);  // DMA next chunk while we compute

    // A fragments (16x32 bf16): lane half h -> K in {8h..8h+7, 16+8h..16+8h+7}
    bf16x16 ahi[2], alo[2];
    #pragma unroll
    for (int mi = 0; mi < 2; ++mi) {
      int row = 32 * wm + 16 * mi + r;
      ahi[mi] = concat8(*(const bf16x8*)&Xhi[row][8 * h],
                        *(const bf16x8*)&Xhi[row][16 + 8 * h]);
      alo[mi] = concat8(*(const bf16x8*)&Xlo[row][8 * h],
                        *(const bf16x8*)&Xlo[row][16 + 8 * h]);
    }

    // B fragments (32x16 bf16): lane half h -> K in {8h..8h+7, 16+8h..16+8h+7}, col = r
    #pragma unroll
    for (int ni = 0; ni < 4; ++ni) {
      int col = 64 * wn + 16 * ni + r;
      bf16x16 bhi = concat8(*(const bf16x8*)&Mhi[col][8 * h],
                            *(const bf16x8*)&Mhi[col][16 + 8 * h]);
      bf16x16 blo = concat8(*(const bf16x8*)&Mlo[col][8 * h],
                            *(const bf16x8*)&Mlo[col][16 + 8 * h]);
      #pragma unroll
      for (int mi = 0; mi < 2; ++mi) {
        acc[mi][ni] = wmma_bf16(ahi[mi], bhi, acc[mi][ni]);
        acc[mi][ni] = wmma_bf16(ahi[mi], blo, acc[mi][ni]);
        acc[mi][ni] = wmma_bf16(alo[mi], bhi, acc[mi][ni]);
      }
    }

    if (more) {
      wait_async();       // raw tiles for kc+32 landed
      __syncthreads();    // all waves done with current bf16 tiles
      convert_tiles();    // raw -> bf16 (overwrites compute tiles)
      __syncthreads();
    }
  }

  // ---- fused epilogue: exp(-g*(xsq - 2*cross + musq)) ----
  // C/D layout: VGPR j, lane (h,r) -> M = j + 8h, N = r
  #pragma unroll
  for (int mi = 0; mi < 2; ++mi) {
    float xs[8];
    #pragma unroll
    for (int j = 0; j < 8; ++j)
      xs[j] = xsq[m0 + 32 * wm + 16 * mi + 8 * h + j];
    #pragma unroll
    for (int ni = 0; ni < 4; ++ni) {
      int   col = n0 + 64 * wn + 16 * ni + r;
      float mq  = musq[col];
      #pragma unroll
      for (int j = 0; j < 8; ++j) {
        int   row = m0 + 32 * wm + 16 * mi + 8 * h + j;
        float l2  = xs[j] - 2.0f * acc[mi][ni][j] + mq;
        out[(size_t)row * UNITS + col] = __expf(-GAMMA * l2);
      }
    }
  }
}

extern "C" void kernel_launch(void* const* d_in, const int* in_sizes, int n_in,
                              void* d_out, int out_size, void* d_ws, size_t ws_size,
                              hipStream_t stream) {
  const float* X  = (const float*)d_in[0];   // [16384, 512]
  const float* MU = (const float*)d_in[1];   // [512, 1024]
  float* out  = (float*)d_out;               // [16384, 1024]
  float* xsq  = (float*)d_ws;                // [16384]
  float* musq = xsq + BATCH;                 // [1024]

  xsq_kernel<<<(BATCH * 32 + 255) / 256, 256, 0, stream>>>(X, xsq);
  musq_kernel<<<(UNITS + 255) / 256, 256, 0, stream>>>(MU, musq);

  dim3 grid(UNITS / 128, BATCH / 128);  // (8, 128)
  rbf_wmma_kernel<<<grid, 256, 0, stream>>>(X, MU, xsq, musq, out);
}
